// N3GCN_6098853560424
// MI455X (gfx1250) — matrix operand order repaired and checked by reading, not compiled, not run on hardware
//
#include <hip/hip_runtime.h>
#include <hip/hip_bf16.h>
#include <cstdint>
#include <cstddef>

typedef __attribute__((ext_vector_type(16))) __bf16 v16bf;
typedef __attribute__((ext_vector_type(8)))  float  v8f;

#define NN    8192
#define EE    65536
#define F_IN  512
#define H1D   4096
#define H2D   1024
#define F_OUT 10

// ---------------- elementwise / prep kernels ----------------

__global__ void f2bf_kernel(const float* __restrict__ in, __bf16* __restrict__ out, int n) {
  int i = blockIdx.x * blockDim.x + threadIdx.x;
  if (i < n) out[i] = (__bf16)in[i];
}

// in: [K][N] fp32 row-major  ->  out: [N][K] bf16 row-major (transposed cast)
__global__ __launch_bounds__(256) void f2bf_transpose_kernel(
    const float* __restrict__ in, __bf16* __restrict__ out, int K, int N) {
  __shared__ float tile[32][33];
  int kb = blockIdx.x * 32, nb = blockIdx.y * 32;
  int tx = threadIdx.x & 31, ty = threadIdx.x >> 5;  // 8 rows of 32
#pragma unroll
  for (int i = ty; i < 32; i += 8)
    tile[i][tx] = in[(size_t)(kb + i) * N + (nb + tx)];
  __syncthreads();
#pragma unroll
  for (int i = ty; i < 32; i += 8)
    out[(size_t)(nb + i) * K + (kb + tx)] = (__bf16)tile[tx][i];
}

__global__ void deg_init_kernel(float* __restrict__ deg, int n) {
  int i = blockIdx.x * blockDim.x + threadIdx.x;
  if (i < n) deg[i] = 1.0f;  // self-loop contribution
}

__global__ void deg_accum_kernel(const long long* __restrict__ dst, float* __restrict__ deg, int E) {
  int i = blockIdx.x * blockDim.x + threadIdx.x;
  if (i < E) atomicAdd(&deg[(int)dst[i]], 1.0f);
}

__device__ __forceinline__ float dinv_sqrt(float d) {
  return d > 0.0f ? rsqrtf(fmaxf(d, 1.0f)) : 0.0f;
}

__global__ void norm_kernel(const long long* __restrict__ src, const long long* __restrict__ dst,
                            const float* __restrict__ deg, float* __restrict__ norm, int E) {
  int i = blockIdx.x * blockDim.x + threadIdx.x;
  if (i < E) norm[i] = dinv_sqrt(deg[(int)src[i]]) * dinv_sqrt(deg[(int)dst[i]]);
}

__global__ void snorm_kernel(const float* __restrict__ deg, float* __restrict__ sn, int n) {
  int i = blockIdx.x * blockDim.x + threadIdx.x;
  if (i < n) { float di = dinv_sqrt(deg[i]); sn[i] = di * di; }
}

// ---------------- bf16 WMMA GEMM: C[M,N] = A[M,K] x Bt[N,K]^T ----------------
// Block tile 128x256, K-step 32; 8 waves as 2(M) x 4(N), 64x64 per wave
// (4x4 WMMA accumulators -> 16 v_wmma per k-step per wave vs 16 ds_load_b128).
// Tile feed: double-buffered global_load_async_to_lds_b128 (ASYNCcnt) ping-pong.
// Epilogue writes both H and AGG = H * snrm[m] (fused GCN self-loop term).

union FragU {
  uint4 q[2];
  v16bf v;
};

template <int CH>
__device__ __forceinline__ void tile_async_load(const __bf16* __restrict__ g, int ld,
                                                int row0, int k0, uint32_t ldsBase, int tid) {
  // rows x 32 bf16 tile, 16B chunks: 4 chunks per row; CH chunks per thread.
#pragma unroll
  for (int c = 0; c < CH; ++c) {
    int chunk = tid * CH + c;
    int row = chunk >> 2;
    int col = (chunk & 3) * 8;
    uint32_t loff = ldsBase + (uint32_t)(row * 80 + col * 2);  // padded stride 40 elems
    unsigned long long ga =
        (unsigned long long)(uintptr_t)(g + (size_t)(row0 + row) * ld + (k0 + col));
    asm volatile("global_load_async_to_lds_b128 %0, %1, off"
                 :: "v"(loff), "v"(ga)
                 : "memory");
  }
}

__device__ __forceinline__ void wait_async_all() {
  asm volatile("s_wait_asynccnt 0" ::: "memory");
}

__global__ __launch_bounds__(256) void gemm_bf16_wmma_kernel(
    const __bf16* __restrict__ A, const __bf16* __restrict__ Bt,
    const float* __restrict__ snrm,
    float* __restrict__ C, float* __restrict__ AGG, int M, int N, int K) {
  constexpr int BM = 128, BN = 256, BK = 32;
  constexpr int LDT = 40;  // padded row stride (elems): 80B rows, 16B aligned
  __shared__ __bf16 As[2][BM][LDT];   // 20 KB
  __shared__ __bf16 Bs[2][BN][LDT];   // 40 KB

  const int tid   = threadIdx.x;
  const int lane  = tid & 31;
  const int wave  = tid >> 5;
  const int l16   = lane & 15;
  const int lhalf = lane >> 4;
  const int bm = blockIdx.x * BM;
  const int bn = blockIdx.y * BN;
  const int wm0 = (wave >> 2) * 64;   // 2 waves along M
  const int wn0 = (wave & 3) * 64;    // 4 waves along N

  const uint32_t aBase[2] = {(uint32_t)(uintptr_t)&As[0][0][0],
                             (uint32_t)(uintptr_t)&As[1][0][0]};
  const uint32_t bBase[2] = {(uint32_t)(uintptr_t)&Bs[0][0][0],
                             (uint32_t)(uintptr_t)&Bs[1][0][0]};

  const v8f vzero = {0.f, 0.f, 0.f, 0.f, 0.f, 0.f, 0.f, 0.f};
  v8f acc[4][4];
#pragma unroll
  for (int i = 0; i < 4; ++i)
#pragma unroll
    for (int j = 0; j < 4; ++j) acc[i][j] = vzero;

  // prologue: async-fill buffer 0
  tile_async_load<2>(A,  K, bm, 0, aBase[0], tid);
  tile_async_load<4>(Bt, K, bn, 0, bBase[0], tid);
  wait_async_all();
  __syncthreads();

  const int nk = K / BK;
  for (int kb = 0; kb < nk; ++kb) {
    const int cur = kb & 1;
    if (kb + 1 < nk) {  // prefetch next tile into the other buffer (async DMA)
      tile_async_load<2>(A,  K, bm, (kb + 1) * BK, aBase[cur ^ 1], tid);
      tile_async_load<4>(Bt, K, bn, (kb + 1) * BK, bBase[cur ^ 1], tid);
    }

    // A fragment (16x32 bf16): lane(l16,h) holds K in [8h,8h+8) and [16+8h,16+8h+8)
    FragU afrag[4];
#pragma unroll
    for (int mf = 0; mf < 4; ++mf) {
      const __bf16* p = &As[cur][wm0 + mf * 16 + l16][0];
      afrag[mf].q[0] = *(const uint4*)(p + lhalf * 8);
      afrag[mf].q[1] = *(const uint4*)(p + 16 + lhalf * 8);
    }
    // B fragment (32x16 bf16): lane(l16,h) holds K in [8h,8h+8) and [16+8h,16+8h+8)
    FragU bfrag[4];
#pragma unroll
    for (int nf = 0; nf < 4; ++nf) {
      const __bf16* p = &Bs[cur][wn0 + nf * 16 + l16][0];
      bfrag[nf].q[0] = *(const uint4*)(p + lhalf * 8);
      bfrag[nf].q[1] = *(const uint4*)(p + 16 + lhalf * 8);
    }

#pragma unroll
    for (int mf = 0; mf < 4; ++mf)
#pragma unroll
      for (int nf = 0; nf < 4; ++nf)
        acc[mf][nf] = __builtin_amdgcn_wmma_f32_16x16x32_bf16(
            false, afrag[mf].v, false, bfrag[nf].v, (short)0, acc[mf][nf],
            false, false);

    wait_async_all();   // next tile fully in LDS
    __syncthreads();    // all waves done reading current tile
  }

  // C/D layout: VGPR r -> lanes 0-15: (M=r, N=lane); lanes 16-31: (M=8+r, N=lane-16)
  // Fused epilogue: H = C; AGG = C * snrm[m]  (GCN self-loop term)
#pragma unroll
  for (int mf = 0; mf < 4; ++mf)
#pragma unroll
    for (int r = 0; r < 8; ++r) {
      int m = bm + wm0 + mf * 16 + lhalf * 8 + r;
      float sn = snrm[m];
#pragma unroll
      for (int nf = 0; nf < 4; ++nf) {
        int n = bn + wn0 + nf * 16 + l16;
        float c = acc[mf][nf][r];
        C[(size_t)m * N + n]   = c;
        AGG[(size_t)m * N + n] = c * sn;
      }
    }
}

// ---------------- small GEMM for layer 3 (N=10) ----------------
__global__ void gemm_small_kernel(const __bf16* __restrict__ A, const __bf16* __restrict__ B,
                                  float* __restrict__ C, int M, int N, int K) {
  int idx = blockIdx.x * blockDim.x + threadIdx.x;
  if (idx >= M * N) return;
  int row = idx / N, col = idx % N;
  float s = 0.0f;
  const __bf16* a = A + (size_t)row * K;
  for (int k = 0; k < K; ++k) s += (float)a[k] * (float)B[(size_t)k * N + col];
  C[idx] = s;
}

// ---------------- graph aggregation ----------------

__global__ void self_term_kernel(const float* __restrict__ h, const float* __restrict__ sn,
                                 float* __restrict__ out, int Nn, int F) {
  int idx = blockIdx.x * blockDim.x + threadIdx.x;
  if (idx < Nn * F) out[idx] = h[idx] * sn[idx / F];
}

__global__ void scatter_vec4_kernel(const float* __restrict__ h,
                                    const long long* __restrict__ srcI,
                                    const long long* __restrict__ dstI,
                                    const float* __restrict__ norm,
                                    float* __restrict__ out, int E, int F) {
  int f = (blockIdx.x * blockDim.x + threadIdx.x) * 4;
  if (f >= F) return;
  int e = blockIdx.y;
  int s = (int)srcI[e], d = (int)dstI[e];
  float w = norm[e];
  float4 hv = *(const float4*)(h + (size_t)s * F + f);
  float* o = out + (size_t)d * F + f;
  atomicAdd(o + 0, hv.x * w);
  atomicAdd(o + 1, hv.y * w);
  atomicAdd(o + 2, hv.z * w);
  atomicAdd(o + 3, hv.w * w);
}

__global__ void scatter_scalar_kernel(const float* __restrict__ h,
                                      const long long* __restrict__ srcI,
                                      const long long* __restrict__ dstI,
                                      const float* __restrict__ norm,
                                      float* __restrict__ out, int E, int F) {
  int idx = blockIdx.x * blockDim.x + threadIdx.x;
  if (idx >= E * F) return;
  int e = idx / F, f = idx % F;
  int s = (int)srcI[e], d = (int)dstI[e];
  atomicAdd(&out[(size_t)d * F + f], h[(size_t)s * F + f] * norm[e]);
}

__global__ void bias_relu_cast_kernel(const float* __restrict__ agg, const float* __restrict__ b,
                                      __bf16* __restrict__ out, int Nn, int F) {
  int idx = blockIdx.x * blockDim.x + threadIdx.x;
  if (idx < Nn * F) out[idx] = (__bf16)fmaxf(agg[idx] + b[idx % F], 0.0f);
}

__global__ void bias_logsoftmax_kernel(const float* __restrict__ agg, const float* __restrict__ b,
                                       float* __restrict__ out, int Nn, int F) {
  int row = blockIdx.x * blockDim.x + threadIdx.x;
  if (row >= Nn) return;
  float v[16];
  float mx = -3.4e38f;
  for (int f = 0; f < F; ++f) { v[f] = agg[(size_t)row * F + f] + b[f]; mx = fmaxf(mx, v[f]); }
  float s = 0.0f;
  for (int f = 0; f < F; ++f) s += __expf(v[f] - mx);
  float l = __logf(s);
  for (int f = 0; f < F; ++f) out[(size_t)row * F + f] = v[f] - mx - l;
}

// ---------------- host orchestration ----------------

static inline int cdiv_ll(long long a, long long b) { return (int)((a + b - 1) / b); }

extern "C" void kernel_launch(void* const* d_in, const int* in_sizes, int n_in,
                              void* d_out, int out_size, void* d_ws, size_t ws_size,
                              hipStream_t stream) {
  const float* x  = (const float*)d_in[0];
  const float* W1 = (const float*)d_in[1];
  const float* b1 = (const float*)d_in[2];
  const float* W2 = (const float*)d_in[3];
  const float* b2 = (const float*)d_in[4];
  const float* W3 = (const float*)d_in[5];
  const float* b3 = (const float*)d_in[6];
  const long long* ei  = (const long long*)d_in[7];
  const long long* src = ei;
  const long long* dst = ei + EE;
  float* out = (float*)d_out;

  char* ws = (char*)d_ws;
  __bf16* xb   = (__bf16*)(ws + (0ull   << 20));  //  8 MB
  __bf16* w1t  = (__bf16*)(ws + (16ull  << 20));  //  4 MB  [H1D][F_IN]
  __bf16* w2t  = (__bf16*)(ws + (24ull  << 20));  //  8 MB  [H2D][H1D]
  __bf16* w3b  = (__bf16*)(ws + (40ull  << 20));  // 20 KB  [H2D][F_OUT] (row-major)
  float*  deg  = (float*) (ws + (41ull  << 20));
  float*  norm = (float*) (ws + (42ull  << 20));
  float*  snrm = (float*) (ws + (43ull  << 20));
  float*  H    = (float*) (ws + (64ull  << 20));  // GEMM out, 128 MB max
  float*  AGG  = (float*) (ws + (192ull << 20));  // aggregation out, 128 MB max
  __bf16* HB   = (__bf16*)(ws + (320ull << 20));  // bf16 activations, 64 MB max

  // casts to bf16 (weights: transposed so GEMM B operand is K-contiguous)
  f2bf_kernel<<<cdiv_ll((long long)NN * F_IN, 256), 256, 0, stream>>>(x, xb, NN * F_IN);
  f2bf_transpose_kernel<<<dim3(F_IN / 32, H1D / 32), 256, 0, stream>>>(W1, w1t, F_IN, H1D);
  f2bf_transpose_kernel<<<dim3(H1D / 32, H2D / 32), 256, 0, stream>>>(W2, w2t, H1D, H2D);
  f2bf_kernel<<<cdiv_ll((long long)H2D * F_OUT, 256), 256, 0, stream>>>(W3, w3b, H2D * F_OUT);

  // degree + symmetric normalization
  deg_init_kernel<<<cdiv_ll(NN, 256), 256, 0, stream>>>(deg, NN);
  deg_accum_kernel<<<cdiv_ll(EE, 256), 256, 0, stream>>>(dst, deg, EE);
  norm_kernel<<<cdiv_ll(EE, 256), 256, 0, stream>>>(src, dst, deg, norm, EE);
  snorm_kernel<<<cdiv_ll(NN, 256), 256, 0, stream>>>(deg, snrm, NN);

  // ---- layer 1: 8192x512 @ 512x4096 ----
  gemm_bf16_wmma_kernel<<<dim3(NN / 128, H1D / 256), 256, 0, stream>>>(
      xb, w1t, snrm, H, AGG, NN, H1D, F_IN);
  scatter_vec4_kernel<<<dim3(cdiv_ll(H1D / 4, 256), EE), 256, 0, stream>>>(H, src, dst, norm, AGG, EE, H1D);
  bias_relu_cast_kernel<<<cdiv_ll((long long)NN * H1D, 256), 256, 0, stream>>>(AGG, b1, HB, NN, H1D);

  // ---- layer 2: 8192x4096 @ 4096x1024 ----
  gemm_bf16_wmma_kernel<<<dim3(NN / 128, H2D / 256), 256, 0, stream>>>(
      HB, w2t, snrm, H, AGG, NN, H2D, H1D);
  scatter_vec4_kernel<<<dim3(cdiv_ll(H2D / 4, 256), EE), 256, 0, stream>>>(H, src, dst, norm, AGG, EE, H2D);
  bias_relu_cast_kernel<<<cdiv_ll((long long)NN * H2D, 256), 256, 0, stream>>>(AGG, b2, HB, NN, H2D);

  // ---- layer 3: 8192x1024 @ 1024x10, then log_softmax ----
  gemm_small_kernel<<<cdiv_ll((long long)NN * F_OUT, 256), 256, 0, stream>>>(HB, w3b, H, NN, F_OUT, H2D);
  self_term_kernel<<<cdiv_ll(NN * F_OUT, 256), 256, 0, stream>>>(H, snrm, AGG, NN, F_OUT);
  scatter_scalar_kernel<<<cdiv_ll((long long)EE * F_OUT, 256), 256, 0, stream>>>(H, src, dst, norm, AGG, EE, F_OUT);
  bias_logsoftmax_kernel<<<cdiv_ll(NN, 256), 256, 0, stream>>>(AGG, b3, out, NN, F_OUT);
}
